// MultiHeadAttention_27805618274754
// MI455X (gfx1250) — matrix operand order repaired and checked by reading, not compiled, not run on hardware
//
#include <hip/hip_runtime.h>

#define B_  4
#define S_  1024
#define D_  1024
#define H_  16
#define DH_ 64

typedef __attribute__((ext_vector_type(16))) _Float16 v16h;
typedef __attribute__((ext_vector_type(8)))  _Float16 v8h;
typedef __attribute__((ext_vector_type(8)))  float    v8f;
typedef __attribute__((ext_vector_type(4)))  float    v4f;

union F16Frag { v16h v; v8h h[2]; };

// D = A(16x32 f16) * B(32x16 f16) + C(16x16 f32)
__device__ __forceinline__ v8f wmma_f16(v16h a, v16h b, v8f c) {
    return __builtin_amdgcn_wmma_f32_16x16x32_f16(
        /*neg_a=*/false, a, /*neg_b=*/false, b,
        /*c_mod=*/(short)0, c, /*reuse_a=*/false, /*reuse_b=*/false);
}

// A fragment (16x32, row-major, K contiguous). lane<16: row=lane, K {0..7,16..23};
// lane>=16: row=lane-16, K {8..15,24..31}.  Two 16B loads per lane.
__device__ __forceinline__ v16h load_frag_a_f16(const _Float16* A, int lda, int lane) {
    const int hf = lane >> 4, r = lane & 15;
    const _Float16* p = A + (size_t)r * lda + hf * 8;
    F16Frag f;
    f.h[0] = *(const v8h*)(p);
    f.h[1] = *(const v8h*)(p + 16);
    return f.v;
}

// Same A fragment but source is fp32 (convert in-register).
__device__ __forceinline__ v16h load_frag_a_f32(const float* A, int lda, int lane) {
    const int hf = lane >> 4, r = lane & 15;
    const float* p = A + (size_t)r * lda + hf * 8;
    v4f x0 = *(const v4f*)(p);
    v4f x1 = *(const v4f*)(p + 4);
    v4f x2 = *(const v4f*)(p + 16);
    v4f x3 = *(const v4f*)(p + 20);
    v16h f;
#pragma unroll
    for (int i = 0; i < 4; ++i) {
        f[i]      = (_Float16)x0[i];
        f[4 + i]  = (_Float16)x1[i];
        f[8 + i]  = (_Float16)x2[i];
        f[12 + i] = (_Float16)x3[i];
    }
    return f;
}

// B fragment (32x16, stored as [N][K] i.e. K contiguous per column).
// lane<16: col=lane, K {0..7,16..23}; lane>=16: col=lane-16, K {8..15,24..31}.
__device__ __forceinline__ v16h load_frag_b_f16(const _Float16* Bm, int ldb, int lane) {
    const int hf = lane >> 4, r = lane & 15;
    const _Float16* p = Bm + (size_t)r * ldb + hf * 8;
    F16Frag f;
    f.h[0] = *(const v8h*)(p);
    f.h[1] = *(const v8h*)(p + 16);
    return f.v;
}

// ---- weight conversion/transposition ------------------------------------
// W [H,D,dh] fp32 -> WT [H,dh,D] f16
__global__ void __launch_bounds__(256) convert_w_qkv(const float* __restrict__ W,
                                                     _Float16* __restrict__ WT) {
    int idx = blockIdx.x * 256 + threadIdx.x;
    if (idx >= H_ * D_ * DH_) return;
    int h = idx >> 16;            // D*dh = 65536
    int rem = idx & 65535;
    int d = rem >> 6, e = rem & 63;
    WT[((size_t)h * DH_ + e) * D_ + d] = (_Float16)W[idx];
}

// Wo [F,D] fp32 -> WoT [D,F] f16
__global__ void __launch_bounds__(256) convert_wo(const float* __restrict__ Wo,
                                                  _Float16* __restrict__ WoT) {
    int idx = blockIdx.x * 256 + threadIdx.x;
    if (idx >= D_ * D_) return;
    int f = idx >> 10, d = idx & 1023;
    WoT[(size_t)d * D_ + f] = (_Float16)Wo[idx];
}

// ---- QKV projection: X[b,s,:] @ W[h,:,:] + bias -------------------------
// One wave computes a 32(s) x 64(e) tile for one (b,h,src): 8 WMMA / (2A+4B) loads.
__global__ void __launch_bounds__(32) proj_kernel(
    const float* __restrict__ q, const float* __restrict__ k, const float* __restrict__ v,
    const _Float16* __restrict__ WqT, const _Float16* __restrict__ WkT, const _Float16* __restrict__ WvT,
    const float* __restrict__ bq, const float* __restrict__ bk, const float* __restrict__ bv,
    _Float16* __restrict__ qh, _Float16* __restrict__ kh, _Float16* __restrict__ vhT)
{
    const int lane = threadIdx.x;
    const int hf = lane >> 4, r = lane & 15;
    const int s0 = blockIdx.x * 32;
    const int bh = blockIdx.y;           // b*H + h
    const int b = bh >> 4, h = bh & 15;
    const int z = blockIdx.z;

    const float*    X    = (z == 0) ? q   : (z == 1) ? k   : v;
    const _Float16* WT   = (z == 0) ? WqT : (z == 1) ? WkT : WvT;
    const float*    bias = (z == 0) ? bq  : (z == 1) ? bk  : bv;

    v8f zero = {};
    v8f accL[4] = {zero, zero, zero, zero};
    v8f accH[4] = {zero, zero, zero, zero};

    const float*    Abase = X + ((size_t)b * S_ + s0) * D_;
    const _Float16* Bbase = WT + (size_t)h * DH_ * D_;

    for (int kt = 0; kt < D_ / 32; ++kt) {
        v16h aL = load_frag_a_f32(Abase + kt * 32, D_, lane);
        v16h aH = load_frag_a_f32(Abase + (size_t)16 * D_ + kt * 32, D_, lane);
#pragma unroll
        for (int nt = 0; nt < 4; ++nt) {
            v16h bf = load_frag_b_f16(Bbase + (size_t)(nt * 16) * D_ + kt * 32, D_, lane);
            accL[nt] = wmma_f16(aL, bf, accL[nt]);
            accH[nt] = wmma_f16(aH, bf, accH[nt]);
        }
    }

    if (z < 2) {
        _Float16* dst = (z == 0) ? qh : kh;   // [B,H,S,dh]
#pragma unroll
        for (int nt = 0; nt < 4; ++nt) {
            float bb = bias[h * DH_ + nt * 16 + r];
#pragma unroll
            for (int j = 0; j < 8; ++j) {
                dst[((size_t)bh * S_ + s0 + hf * 8 + j) * DH_ + nt * 16 + r] =
                    (_Float16)(accL[nt][j] + bb);
                dst[((size_t)bh * S_ + s0 + 16 + hf * 8 + j) * DH_ + nt * 16 + r] =
                    (_Float16)(accH[nt][j] + bb);
            }
        }
    } else {
        // vhT [B,H,dh,S]: lane's 8 rows (m) are contiguous in s -> 16B packed stores
#pragma unroll
        for (int nt = 0; nt < 4; ++nt) {
            float bb = bias[h * DH_ + nt * 16 + r];
            v8h pL, pH;
#pragma unroll
            for (int j = 0; j < 8; ++j) {
                pL[j] = (_Float16)(accL[nt][j] + bb);
                pH[j] = (_Float16)(accH[nt][j] + bb);
            }
            _Float16* row = vhT + ((size_t)bh * DH_ + nt * 16 + r) * S_ + s0;
            *(v8h*)(row + hf * 8)      = pL;
            *(v8h*)(row + 16 + hf * 8) = pH;
        }
    }
}

// ---- attention: logits -> softmax -> score (fp32 out) -> ctx ------------
// 4 waves per workgroup share one 16-row tile: waves split t-tiles (logits),
// rows (softmax), and the K dimension (ctx) with an LDS partial reduction.
__global__ void __launch_bounds__(128) attn_kernel(
    const _Float16* __restrict__ qh, const _Float16* __restrict__ kh,
    const _Float16* __restrict__ vhT,
    float* __restrict__ score, _Float16* __restrict__ ctx)
{
    __shared__ float slog[16][1044];     // 66,816 B; also reused for ctx partials
    const int tid  = threadIdx.x;
    const int wave = tid >> 5;           // 0..3
    const int lane = tid & 31;
    const int hf = lane >> 4, r = lane & 15;
    const int s0 = blockIdx.x * 16;
    const int bh = blockIdx.y;           // b*H + h
    const int b = bh >> 4, h = bh & 15;

    // 1) logits = qh . kh^T / 8 ; wave w computes t-tiles [16w, 16w+16)
    const _Float16* Aq = qh + ((size_t)bh * S_ + s0) * DH_;
    v16h a0 = load_frag_a_f16(Aq + 0,  DH_, lane);
    v16h a1 = load_frag_a_f16(Aq + 32, DH_, lane);
    const _Float16* Bk = kh + (size_t)bh * S_ * DH_;
    v8f zero = {};
    for (int t = 0; t < 16; ++t) {
        const int tt = wave * 16 + t;
        v16h b0 = load_frag_b_f16(Bk + (size_t)(tt * 16) * DH_ + 0,  DH_, lane);
        v16h b1 = load_frag_b_f16(Bk + (size_t)(tt * 16) * DH_ + 32, DH_, lane);
        v8f acc = wmma_f16(a0, b0, zero);
        acc = wmma_f16(a1, b1, acc);
#pragma unroll
        for (int j = 0; j < 8; ++j)
            slog[hf * 8 + j][tt * 16 + r] = acc[j] * 0.125f;
    }
    __syncthreads();

    // 2) exact fp32 softmax; wave w handles rows [4w, 4w+4)
    float* srow_out = score + ((size_t)bh * S_ + s0) * S_;
    for (int mm = 0; mm < 4; ++mm) {
        const int m = wave * 4 + mm;
        float mx = -3.0e38f;
        for (int i = lane; i < S_; i += 32) mx = fmaxf(mx, slog[m][i]);
#pragma unroll
        for (int o = 16; o > 0; o >>= 1) mx = fmaxf(mx, __shfl_xor(mx, o));
        float sum = 0.f;
        for (int i = lane; i < S_; i += 32) {
            float e = __expf(slog[m][i] - mx);
            slog[m][i] = e;
            sum += e;
        }
#pragma unroll
        for (int o = 16; o > 0; o >>= 1) sum += __shfl_xor(sum, o);
        float inv = 1.0f / sum;
        for (int i = lane; i < S_; i += 32) {
            float p = slog[m][i] * inv;
            slog[m][i] = p;
            __builtin_nontemporal_store(p, &srow_out[(size_t)m * S_ + i]); // streaming 256MB
        }
    }
    __syncthreads();

    // 3) ctx = score @ vh ; wave w handles K-steps [8w, 8w+8), all 4 n-tiles.
    v8f acc[4] = {zero, zero, zero, zero};
    const _Float16* Bv = vhT + (size_t)bh * DH_ * S_;
    for (int kk = 0; kk < 8; ++kk) {
        const int kt = wave * 8 + kk;
        v16h a;
        const int c0 = kt * 32 + hf * 8;
#pragma unroll
        for (int i = 0; i < 8; ++i) a[i]     = (_Float16)slog[r][c0 + i];
#pragma unroll
        for (int i = 0; i < 8; ++i) a[8 + i] = (_Float16)slog[r][c0 + 16 + i];
#pragma unroll
        for (int nt = 0; nt < 4; ++nt) {
            v16h bf = load_frag_b_f16(Bv + (size_t)(nt * 16) * S_ + kt * 32, S_, lane);
            acc[nt] = wmma_f16(a, bf, acc[nt]);
        }
    }
    __syncthreads();   // all waves done reading slog

    // partial sums: overlay P[4][16][64] (4096 floats) on slog storage
    float* P = &slog[0][0];
#pragma unroll
    for (int nt = 0; nt < 4; ++nt)
#pragma unroll
        for (int j = 0; j < 8; ++j)
            P[((size_t)wave * 16 + hf * 8 + j) * 64 + nt * 16 + r] = acc[nt][j];
    __syncthreads();

    // reduce over waves + store ctx [B,S,H*dh] (16B packed stores)
    {
        const int e0 = tid * 8;                  // 128 threads x 8 = 1024 elems
        const int m = e0 >> 6, n = e0 & 63;
        v8h pack;
#pragma unroll
        for (int i = 0; i < 8; ++i) {
            float s = P[e0 + i] + P[1024 + e0 + i] + P[2048 + e0 + i] + P[3072 + e0 + i];
            pack[i] = (_Float16)s;
        }
        *(v8h*)(ctx + ((size_t)b * S_ + s0 + m) * D_ + h * DH_ + n) = pack;
    }
}

// ---- output projection: ctx @ Wo + bo (32x64 tile per wave) -------------
__global__ void __launch_bounds__(32) outproj_kernel(
    const _Float16* __restrict__ ctx, const _Float16* __restrict__ WoT,
    const float* __restrict__ bo, float* __restrict__ out)
{
    const int lane = threadIdx.x;
    const int hf = lane >> 4, r = lane & 15;
    const int row0 = blockIdx.x * 32;    // rows over B*S
    const int d0   = blockIdx.y * 64;

    v8f zero = {};
    v8f accL[4] = {zero, zero, zero, zero};
    v8f accH[4] = {zero, zero, zero, zero};
    const _Float16* Ab = ctx + (size_t)row0 * D_;

    for (int kt = 0; kt < D_ / 32; ++kt) {
        v16h aL = load_frag_a_f16(Ab + kt * 32, D_, lane);
        v16h aH = load_frag_a_f16(Ab + (size_t)16 * D_ + kt * 32, D_, lane);
#pragma unroll
        for (int nt = 0; nt < 4; ++nt) {
            v16h bf = load_frag_b_f16(WoT + (size_t)(d0 + nt * 16) * D_ + kt * 32, D_, lane);
            accL[nt] = wmma_f16(aL, bf, accL[nt]);
            accH[nt] = wmma_f16(aH, bf, accH[nt]);
        }
    }
#pragma unroll
    for (int nt = 0; nt < 4; ++nt) {
        float bb = bo[d0 + nt * 16 + r];
#pragma unroll
        for (int j = 0; j < 8; ++j) {
            __builtin_nontemporal_store(
                accL[nt][j] + bb,
                &out[((size_t)row0 + hf * 8 + j) * D_ + d0 + nt * 16 + r]);
            __builtin_nontemporal_store(
                accH[nt][j] + bb,
                &out[((size_t)row0 + 16 + hf * 8 + j) * D_ + d0 + nt * 16 + r]);
        }
    }
}

extern "C" void kernel_launch(void* const* d_in, const int* in_sizes, int n_in,
                              void* d_out, int out_size, void* d_ws, size_t ws_size,
                              hipStream_t stream) {
    (void)in_sizes; (void)n_in; (void)out_size; (void)ws_size;

    const float* q  = (const float*)d_in[0];
    const float* k  = (const float*)d_in[1];
    const float* v  = (const float*)d_in[2];
    const float* Wq = (const float*)d_in[3];
    const float* bq = (const float*)d_in[4];
    const float* Wk = (const float*)d_in[5];
    const float* bk = (const float*)d_in[6];
    const float* Wv = (const float*)d_in[7];
    const float* bv = (const float*)d_in[8];
    const float* Wo = (const float*)d_in[9];
    const float* bo = (const float*)d_in[10];

    // workspace layout (256B aligned): 4 x 2MB weights + 3 x 8MB heads + 8MB ctx = 40MB
    char* ws = (char*)d_ws;
    size_t off = 0;
    auto alloc = [&](size_t bytes) {
        void* p = ws + off;
        off += (bytes + 255) & ~(size_t)255;
        return p;
    };
    const size_t wElems  = (size_t)H_ * D_ * DH_;      // 1M
    const size_t hdElems = (size_t)B_ * H_ * S_ * DH_; // 4M

    _Float16* WqT = (_Float16*)alloc(wElems * 2);
    _Float16* WkT = (_Float16*)alloc(wElems * 2);
    _Float16* WvT = (_Float16*)alloc(wElems * 2);
    _Float16* WoT = (_Float16*)alloc((size_t)D_ * D_ * 2);
    _Float16* qh  = (_Float16*)alloc(hdElems * 2);
    _Float16* kh  = (_Float16*)alloc(hdElems * 2);
    _Float16* vhT = (_Float16*)alloc(hdElems * 2);
    _Float16* ctx = (_Float16*)alloc((size_t)B_ * S_ * D_ * 2);

    float* out   = (float*)d_out;
    float* score = out + (size_t)B_ * S_ * D_;   // tuple: (out, score)

    convert_w_qkv<<<4096, 256, 0, stream>>>(Wq, WqT);
    convert_w_qkv<<<4096, 256, 0, stream>>>(Wk, WkT);
    convert_w_qkv<<<4096, 256, 0, stream>>>(Wv, WvT);
    convert_wo  <<<4096, 256, 0, stream>>>(Wo, WoT);

    proj_kernel<<<dim3(S_ / 32, B_ * H_, 3), 32, 0, stream>>>(
        q, k, v, WqT, WkT, WvT, bq, bk, bv, qh, kh, vhT);

    attn_kernel<<<dim3(S_ / 16, B_ * H_), 128, 0, stream>>>(qh, kh, vhT, score, ctx);

    outproj_kernel<<<dim3((B_ * S_) / 32, D_ / 64), 32, 0, stream>>>(ctx, WoT, bo, out);
}
